// GAT_36910948942293
// MI455X (gfx1250) — matrix operand order repaired and checked by reading, not compiled, not run on hardware
//
#include <hip/hip_runtime.h>
#include <hip/hip_bf16.h>

#define HH 4
#define CC 32
#define DIN 128
#define DE 16
#define NEG_SLOPE 0.2f

typedef _Float16 half8  __attribute__((ext_vector_type(8)));
typedef _Float16 half16 __attribute__((ext_vector_type(16)));
typedef float    floatx8 __attribute__((ext_vector_type(8)));

// ---- gfx1250 async global->LDS copy (ASYNCcnt-tracked) ----
__device__ __forceinline__ void g2l_async_b128(const void* g, void* l) {
  unsigned loff = (unsigned)(unsigned long long)l;   // low 32 bits of generic ptr == LDS byte addr
  asm volatile("global_load_async_to_lds_b128 %0, %1, off"
               :: "v"(loff), "v"(g) : "memory");
}
__device__ __forceinline__ void wait_async0() {
  asm volatile("s_wait_asynccnt 0x0" ::: "memory");
}

// ---- monotone float<->uint key for atomic max over signed floats ----
__device__ __forceinline__ unsigned f2key(float f) {
  unsigned b = __float_as_uint(f);
  return (b & 0x80000000u) ? ~b : (b | 0x80000000u);
}
__device__ __forceinline__ float key2f(unsigned k) {
  unsigned b = (k & 0x80000000u) ? (k & 0x7FFFFFFFu) : ~k;
  return __uint_as_float(b);
}

// ---------------- utility kernels ----------------
__global__ void k_zero(float* p, long count) {
  long stride = (long)gridDim.x * blockDim.x;
  for (long i = (long)blockIdx.x * blockDim.x + threadIdx.x; i < count; i += stride)
    p[i] = 0.0f;
}

__global__ void k_f32_to_f16(const float* __restrict__ in, _Float16* __restrict__ out, long count) {
  long i = (long)blockIdx.x * blockDim.x + threadIdx.x;
  if (i < count) out[i] = (_Float16)in[i];
}

// W [128,128] row-major -> Wt16 [col][k] f16
__global__ void k_transposeW(const float* __restrict__ W, _Float16* __restrict__ Wt) {
  int t = blockIdx.x * blockDim.x + threadIdx.x;   // 16384
  if (t >= DIN * DIN) return;
  int col = t / DIN, k = t % DIN;
  Wt[(size_t)col * DIN + k] = (_Float16)W[(size_t)k * DIN + col];
}

// edge_attr raw column sums (for mean)
__global__ void k_meansum(const float* __restrict__ ea, float* __restrict__ msum, int E) {
  __shared__ float s[DE];
  if (threadIdx.x < DE) s[threadIdx.x] = 0.0f;
  __syncthreads();
  int e = blockIdx.x * blockDim.x + threadIdx.x;
  if (e < E) {
    const float* row = ea + (size_t)e * DE;
#pragma unroll
    for (int d = 0; d < DE; ++d) atomicAdd(&s[d], row[d]);
  }
  __syncthreads();
  if (threadIdx.x < DE) atomicAdd(&msum[threadIdx.x], s[threadIdx.x]);
}

// v_edge[d][h] = sum_c W_edge[d, h*C+c] * att_edge[h,c]
__global__ void k_vedge(const float* __restrict__ We, const float* __restrict__ ae,
                        float* __restrict__ vedge) {
  int t = blockIdx.x * blockDim.x + threadIdx.x;   // 64
  if (t >= DE * HH) return;
  int d = t >> 2, h = t & 3;
  float s = 0.0f;
#pragma unroll
  for (int c = 0; c < CC; ++c) s += We[(size_t)d * DIN + h * CC + c] * ae[h * CC + c];
  vedge[d * HH + h] = s;
}

// edge_const[h] = mean_edge . v_edge[:,h]
__global__ void k_edgeconst(const float* __restrict__ msum, const float* __restrict__ vedge,
                            float* __restrict__ econst, float invE) {
  int h = threadIdx.x;
  if (h >= HH) return;
  float s = 0.0f;
#pragma unroll
  for (int d = 0; d < DE; ++d) s += msum[d] * invE * vedge[d * HH + h];
  econst[h] = s;
}

// ---------------- WMMA GEMM with async global->LDS staging ----------------
// H = A16[N,128] x W[128,128]; Bt16 = W^T (f16).
// Persistent blocks: stage full 32KB W^T into LDS once, then loop 16-row tiles,
// staging each 4KB A tile with one async b128 per thread.
__global__ void __launch_bounds__(256) k_gemm_wmma(const _Float16* __restrict__ A16,
                                                   const _Float16* __restrict__ Bt16,
                                                   float* __restrict__ Hout, int nrows) {
  __shared__ _Float16 sB[DIN * DIN];   // 32 KB
  __shared__ _Float16 sA[16 * DIN];    // 4 KB
  const int tid    = threadIdx.x;
  const int lane   = tid & 31;
  const int wave   = tid >> 5;         // 0..7 -> 16-col tile
  const int r      = lane & 15;
  const int hiHalf = lane >> 4;        // 0 or 1
  const int colBase = wave * 16;
  const int nTiles  = nrows >> 4;

  // stage all of B (W^T): 16384 halfs, 256 threads x 8 iters x 16B
#pragma unroll
  for (int it = 0; it < 8; ++it) {
    int idx = (it * 256 + tid) * 8;    // in halfs
    g2l_async_b128(Bt16 + idx, sB + idx);
  }
  wait_async0();
  __syncthreads();

  const _Float16* aRow = sA + (size_t)r * DIN;              // A row for this lane
  const _Float16* bCol = sB + (size_t)(colBase + r) * DIN;  // B column for this lane

  for (int rt = blockIdx.x; rt < nTiles; rt += gridDim.x) {
    const int row0 = rt * 16;
    // stage A tile (16x128 f16 = 4KB): one async 16B copy per thread
    g2l_async_b128(A16 + (size_t)row0 * DIN + tid * 8, sA + tid * 8);
    wait_async0();
    __syncthreads();

    floatx8 acc = {};
#pragma unroll
    for (int k0 = 0; k0 < DIN; k0 += 32) {
      // A (16x32 f16): lanes 0-15 hold K 0..7 & 16..23, lanes 16-31 hold K 8..15 & 24..31
      int ab = k0 + hiHalf * 8;
      half8 a_lo = *(const half8*)(aRow + ab);
      half8 a_hi = *(const half8*)(aRow + ab + 16);
      // B (32x16 f16): lanes 0-15 hold K 0..7 & 16..23, lanes 16-31 hold K 8..15 & 24..31
      int bb = k0 + hiHalf * 8;
      half8 b_lo = *(const half8*)(bCol + bb);
      half8 b_hi = *(const half8*)(bCol + bb + 16);
      half16 a, b;
#pragma unroll
      for (int i = 0; i < 8; ++i) {
        a[i] = a_lo[i]; a[8 + i] = a_hi[i];
        b[i] = b_lo[i]; b[8 + i] = b_hi[i];
      }
      acc = __builtin_amdgcn_wmma_f32_16x16x32_f16(false, a, false, b, (short)0, acc, false, false);
    }
    // C/D layout: col = lane%16, row m = rr + (lane>=16 ? 8 : 0)
    float* outp = Hout + (size_t)row0 * DIN + colBase;
#pragma unroll
    for (int rr = 0; rr < 8; ++rr) {
      int m = rr + (hiHalf ? 8 : 0);
      outp[(size_t)m * DIN + r] = acc[rr];
    }
    __syncthreads();   // all waves done reading sA before next stage overwrites it
  }
}

// ---------------- attention scores s_src[n,h], s_dst[n,h] ----------------
__global__ void k_node_scores(const float* __restrict__ H, const float* __restrict__ as,
                              const float* __restrict__ ad, float* __restrict__ ssrc,
                              float* __restrict__ sdst, int n) {
  long t = (long)blockIdx.x * blockDim.x + threadIdx.x;   // n*8
  long node = t >> 3;
  if (node >= n) return;
  int rem = (int)(t & 7);
  int h = rem & 3;
  bool isDst = rem >= 4;
  const float* av = (isDst ? ad : as) + h * CC;
  const float* hp = H + node * DIN + h * CC;
  float s = 0.0f;
#pragma unroll
  for (int c = 0; c < CC; ++c) s += hp[c] * av[c];
  (isDst ? sdst : ssrc)[node * HH + h] = s;
}

// ---------------- pass 1: logits + leaky relu + atomic max ----------------
__global__ void k_edge_alpha(const int* __restrict__ src, const int* __restrict__ dst,
                             const float* __restrict__ ea, const float* __restrict__ vedge,
                             const float* __restrict__ ssrc, const float* __restrict__ sdst,
                             float* __restrict__ alpha, unsigned* __restrict__ amax, int E) {
  int e = blockIdx.x * blockDim.x + threadIdx.x;
  if (e >= E) return;
  int s = src[e], d = dst[e];
  float dv[HH] = {0.f, 0.f, 0.f, 0.f};
  const float* row = ea + (size_t)e * DE;
#pragma unroll
  for (int k = 0; k < DE; ++k) {
    float v = row[k];
#pragma unroll
    for (int h = 0; h < HH; ++h) dv[h] += v * vedge[k * HH + h];
  }
#pragma unroll
  for (int h = 0; h < HH; ++h) {
    float a = ssrc[(size_t)s * HH + h] + sdst[(size_t)d * HH + h] + dv[h];
    a = a > 0.0f ? a : NEG_SLOPE * a;
    alpha[(size_t)e * HH + h] = a;
    atomicMax(&amax[(size_t)d * HH + h], f2key(a));
  }
}

__global__ void k_loop_alpha(const float* __restrict__ ssrc, const float* __restrict__ sdst,
                             const float* __restrict__ econst, float* __restrict__ alphaL,
                             unsigned* __restrict__ amax, int n) {
  int node = blockIdx.x * blockDim.x + threadIdx.x;
  if (node >= n) return;
#pragma unroll
  for (int h = 0; h < HH; ++h) {
    float a = ssrc[(size_t)node * HH + h] + sdst[(size_t)node * HH + h] + econst[h];
    a = a > 0.0f ? a : NEG_SLOPE * a;
    alphaL[(size_t)node * HH + h] = a;
    atomicMax(&amax[(size_t)node * HH + h], f2key(a));
  }
}

// ---------------- pass 2: exp + denom ----------------
__global__ void k_edge_exp(const int* __restrict__ dst, float* __restrict__ alpha,
                           const unsigned* __restrict__ amax, float* __restrict__ denom, int E) {
  int e = blockIdx.x * blockDim.x + threadIdx.x;
  if (e >= E) return;
  int d = dst[e];
#pragma unroll
  for (int h = 0; h < HH; ++h) {
    float m = key2f(amax[(size_t)d * HH + h]);
    float ex = __expf(alpha[(size_t)e * HH + h] - m);
    alpha[(size_t)e * HH + h] = ex;
    atomicAdd(&denom[(size_t)d * HH + h], ex);
  }
}

__global__ void k_loop_exp(float* __restrict__ alphaL, const unsigned* __restrict__ amax,
                           float* __restrict__ denom, int n) {
  int node = blockIdx.x * blockDim.x + threadIdx.x;
  if (node >= n) return;
#pragma unroll
  for (int h = 0; h < HH; ++h) {
    float m = key2f(amax[(size_t)node * HH + h]);
    float ex = __expf(alphaL[(size_t)node * HH + h] - m);
    alphaL[(size_t)node * HH + h] = ex;
    atomicAdd(&denom[(size_t)node * HH + h], ex);
  }
}

// ---------------- pass 3: weighted scatter-aggregate (one wave per edge) ----------------
__global__ void k_edge_agg(const int* __restrict__ src, const int* __restrict__ dst,
                           const float* __restrict__ ex, const float* __restrict__ denom,
                           const float* __restrict__ H, float* __restrict__ outacc, int E) {
  long t = (long)blockIdx.x * blockDim.x + threadIdx.x;
  long e = t >> 5;
  if (e >= E) return;
  int lane = (int)(t & 31);
  int off = lane * 4;           // 4 floats per lane -> 128 per edge
  int h = off >> 5;             // head index
  int s = src[e], d = dst[e];
  float coef = ex[e * HH + h] / (denom[(size_t)d * HH + h] + 1e-16f);
  const float4 v = *(const float4*)(H + (size_t)s * DIN + off);
  float* op = outacc + (size_t)d * DIN + off;
  atomicAdd(&op[0], v.x * coef);
  atomicAdd(&op[1], v.y * coef);
  atomicAdd(&op[2], v.z * coef);
  atomicAdd(&op[3], v.w * coef);
}

__global__ void k_loop_agg(const float* __restrict__ exL, const float* __restrict__ denom,
                           const float* __restrict__ H, float* __restrict__ outacc, int n) {
  long t = (long)blockIdx.x * blockDim.x + threadIdx.x;
  long node = t >> 5;
  if (node >= n) return;
  int lane = (int)(t & 31);
  int off = lane * 4;
  int h = off >> 5;
  float coef = exL[node * HH + h] / (denom[node * HH + h] + 1e-16f);
  const float4 v = *(const float4*)(H + node * DIN + off);
  float* op = outacc + node * DIN + off;
  atomicAdd(&op[0], v.x * coef);
  atomicAdd(&op[1], v.y * coef);
  atomicAdd(&op[2], v.z * coef);
  atomicAdd(&op[3], v.w * coef);
}

// ---------------- finalize ----------------
__global__ void k_finalize_concat(const float* __restrict__ outacc, const float* __restrict__ bias,
                                  _Float16* __restrict__ xnext, long count) {
  long i = (long)blockIdx.x * blockDim.x + threadIdx.x;
  if (i >= count) return;
  xnext[i] = (_Float16)(outacc[i] + bias[i & (DIN - 1)]);
}

__global__ void k_finalize_last(const float* __restrict__ outacc, const float* __restrict__ bias,
                                float* __restrict__ dout, int n) {
  long t = (long)blockIdx.x * blockDim.x + threadIdx.x;
  long node = t >> 5;
  if (node >= n) return;
  int c = (int)(t & 31);
  const float* row = outacc + node * DIN;
  float m = 0.25f * (row[c] + row[CC + c] + row[2 * CC + c] + row[3 * CC + c]);
  dout[node * CC + c] = m + bias[c];
}

// ---------------- host launch ----------------
extern "C" void kernel_launch(void* const* d_in, const int* in_sizes, int n_in,
                              void* d_out, int out_size, void* d_ws, size_t ws_size,
                              hipStream_t stream) {
  const float* x         = (const float*)d_in[0];
  const float* edge_attr = (const float*)d_in[1];
  const int*   edge_idx  = (const int*)d_in[2];
  const float* W         = (const float*)d_in[3];
  const float* att_src   = (const float*)d_in[4];
  const float* att_dst   = (const float*)d_in[5];
  const float* W_edge    = (const float*)d_in[6];
  const float* att_edge  = (const float*)d_in[7];
  const float* bias_cat  = (const float*)d_in[8];
  const float* bias_last = (const float*)d_in[9];
  float* dout = (float*)d_out;

  const int n = in_sizes[0] / DIN;     // 50000 (multiple of 16)
  const int E = in_sizes[1] / DE;      // 800000
  const int L = in_sizes[3] / (DIN * DIN);   // 3
  const int* srcIdx = edge_idx;
  const int* dstIdx = edge_idx + E;

  // workspace carve-up (256B aligned)
  char* base = (char*)d_ws;
  size_t off = 0;
  auto alloc = [&](size_t bytes) -> char* {
    char* p = base + off;
    off += (bytes + 255) & ~(size_t)255;
    return p;
  };
  _Float16* x16    = (_Float16*)alloc((size_t)n * DIN * sizeof(_Float16));
  _Float16* wt16   = (_Float16*)alloc((size_t)DIN * DIN * sizeof(_Float16));
  float*    Hbuf   = (float*)alloc((size_t)n * DIN * sizeof(float));
  float*    ssrc   = (float*)alloc((size_t)n * HH * sizeof(float));
  float*    sdst   = (float*)alloc((size_t)n * HH * sizeof(float));
  unsigned* amax   = (unsigned*)alloc((size_t)n * HH * sizeof(unsigned));
  float*    denom  = (float*)alloc((size_t)n * HH * sizeof(float));
  float*    aE     = (float*)alloc((size_t)E * HH * sizeof(float));
  float*    aL     = (float*)alloc((size_t)n * HH * sizeof(float));
  float*    outacc = (float*)alloc((size_t)n * DIN * sizeof(float));
  float*    msum   = (float*)alloc(DE * sizeof(float));
  float*    vedge  = (float*)alloc(DE * HH * sizeof(float));
  float*    econst = (float*)alloc(HH * sizeof(float));

  const int B = 256;
  auto cdiv = [](long a, long b) { return (unsigned)((a + b - 1) / b); };

  const int nTiles = n / 16;
  const unsigned gemmGrid = (unsigned)(nTiles < 625 ? nTiles : 625);

  // one-time prep: mean of edge_attr columns; x -> f16
  k_zero<<<1, DE, 0, stream>>>(msum, DE);
  k_meansum<<<cdiv(E, B), B, 0, stream>>>(edge_attr, msum, E);
  k_f32_to_f16<<<cdiv((long)n * DIN, B), B, 0, stream>>>(x, x16, (long)n * DIN);

  for (int i = 0; i < L; ++i) {
    const float* Wi  = W + (size_t)i * DIN * DIN;
    const float* asi = att_src + (size_t)i * HH * CC;
    const float* adi = att_dst + (size_t)i * HH * CC;
    const float* Wei = W_edge + (size_t)i * DE * DIN;
    const float* aei = att_edge + (size_t)i * HH * CC;

    // small per-layer precompute
    k_transposeW<<<cdiv(DIN * DIN, B), B, 0, stream>>>(Wi, wt16);
    k_vedge<<<1, 64, 0, stream>>>(Wei, aei, vedge);
    k_edgeconst<<<1, HH, 0, stream>>>(msum, vedge, econst, 1.0f / (float)E);

    // dense GEMM on WMMA with async LDS staging: H = x16 @ W
    k_gemm_wmma<<<gemmGrid, 256, 0, stream>>>(x16, wt16, Hbuf, n);

    // attention scores
    k_node_scores<<<cdiv((long)n * 8, B), B, 0, stream>>>(Hbuf, asi, adi, ssrc, sdst, n);

    // reset per-layer accumulators (amax key 0 == below any real key)
    k_zero<<<cdiv((long)n * HH, B), B, 0, stream>>>((float*)amax, (long)n * HH);
    k_zero<<<cdiv((long)n * HH, B), B, 0, stream>>>(denom, (long)n * HH);
    k_zero<<<cdiv((long)n * DIN, B), B, 0, stream>>>(outacc, (long)n * DIN);

    // scatter softmax
    k_edge_alpha<<<cdiv(E, B), B, 0, stream>>>(srcIdx, dstIdx, edge_attr, vedge, ssrc, sdst, aE, amax, E);
    k_loop_alpha<<<cdiv(n, B), B, 0, stream>>>(ssrc, sdst, econst, aL, amax, n);
    k_edge_exp<<<cdiv(E, B), B, 0, stream>>>(dstIdx, aE, amax, denom, E);
    k_loop_exp<<<cdiv(n, B), B, 0, stream>>>(aL, amax, denom, n);
    k_edge_agg<<<cdiv((long)E * 32, B), B, 0, stream>>>(srcIdx, dstIdx, aE, denom, Hbuf, outacc, E);
    k_loop_agg<<<cdiv((long)n * 32, B), B, 0, stream>>>(aL, denom, Hbuf, outacc, n);

    if (i != L - 1) {
      const float* bi = bias_cat + (size_t)i * DIN;
      k_finalize_concat<<<cdiv((long)n * DIN, B), B, 0, stream>>>(outacc, bi, x16, (long)n * DIN);
    } else {
      k_finalize_last<<<cdiv((long)n * 32, B), B, 0, stream>>>(outacc, bias_last, dout, n);
    }
  }
}
